// MultiHeadAttention_14568529068187
// MI455X (gfx1250) — compile-verified, matrix operand order checked by reading, not run on hardware
//
#include <hip/hip_runtime.h>
#include <hip/hip_bf16.h>

// ---------------------------------------------------------------------------
// MHA forward for MI455X (gfx1250): wave32, WMMA bf16 (f32 accumulate),
// all tile staging via Tensor Data Mover (tensor_load_to_lds), double-buffered.
// Transposes hoisted out of hot loops (pre-transposed weights, V stored [hd][s]).
// B=2, S=2048, D=1024, H=16, HD=64.
// ---------------------------------------------------------------------------

typedef __attribute__((ext_vector_type(16))) __bf16        v16bf;
typedef __attribute__((ext_vector_type(8)))  float         v8f;
typedef __attribute__((ext_vector_type(4)))  unsigned int  v4u;
typedef __attribute__((ext_vector_type(8)))  int           v8i;
typedef __attribute__((ext_vector_type(4)))  int           v4i;

#define S_LEN 2048
#define D_DIM 1024
#define NH    16
#define HD    64
#define NB    2

#if __has_builtin(__builtin_amdgcn_tensor_load_to_lds)
#define HAVE_TDM 1
#else
#define HAVE_TDM 0
#endif

__device__ __forceinline__ v8f wmma_bf16(v16bf a, v16bf b, v8f c) {
  // D = A(16x32 bf16) * B(32x16 bf16) + C(16x16 f32)
  return __builtin_amdgcn_wmma_f32_16x16x32_bf16(false, a, false, b,
                                                 (short)0, c, false, false);
}

// 16x32 bf16 fragment load (A layout; also B layout when tile stored [n][k]).
// ISA 7.12.2: lanes 0-15 row=lane, VGPR0-3 K=0..7 pairs, VGPR4-7 K=16..23;
// lanes 16-31 same rows, K offset +8. stride/k_off must be even.
__device__ __forceinline__ v16bf load_frag(const __bf16* tile, int row_off,
                                           int stride, int k_off) {
  const int lane  = threadIdx.x & 31;
  const int r     = row_off + (lane & 15);
  const int kbase = (lane >> 4) << 3;  // 0 or 8
  const unsigned int* p32 =
      (const unsigned int*)(tile + r * stride + k_off + kbase);
  union { v16bf v; unsigned int u[8]; } f;
#pragma unroll
  for (int i = 0; i < 4; ++i) {
    f.u[i]     = p32[i];      // K = kbase + 2i
    f.u[4 + i] = p32[8 + i];  // K = 16 + kbase + 2i
  }
  return f.v;
}

#if HAVE_TDM
// Loop-invariant part of a TDM descriptor (group 1): tile dims, global row
// stride, data size, and LDS row padding. D# packing per CDNA5 ISA ch.8.
__device__ __forceinline__ v8i tdm_make_g1(unsigned tile_w, unsigned tile_h,
                                           unsigned long long stride_elems,
                                           unsigned pad_interval_code,
                                           unsigned pad_amount_code) {
  v8i g1;
  g1[0] = (int)((1u << 16)                        // data_size = 2 bytes
                | (1u << 20)                      // pad_enable
                | (pad_interval_code << 22) | (pad_amount_code << 25));
  g1[1] = (int)((tile_w & 0xFFFFu) << 16);        // tensor_dim0[15:0]
  g1[2] = (int)((tile_w >> 16) | ((tile_h & 0xFFFFu) << 16));  // dim0 hi|dim1 lo
  g1[3] = (int)((tile_h >> 16) | (tile_w << 16)); // dim1 hi | tile_dim0
  g1[4] = (int)(tile_h);                          // tile_dim1 (tile_dim2 = 0)
  g1[5] = (int)(stride_elems & 0xFFFFFFFFull);    // tensor_dim0_stride lo
  g1[6] = (int)((stride_elems >> 32) & 0xFFFFull);
  g1[7] = 0;
  return g1;
}

// Per-issue part (group 0): LDS address + global tile address. EXEC ignored;
// issue from one wave, completion via TENSORcnt.
__device__ __forceinline__ void tdm_issue(unsigned lds_addr, const void* gptr,
                                          v8i g1) {
  unsigned long long ga = (unsigned long long)gptr;
  v4u g0 = {1u,                                   // count=1, user mode
            lds_addr,                             // lds_addr [63:32]
            (unsigned)ga,                         // global_addr [95:64]
            (unsigned)((ga >> 32) & 0x01FFFFFFu)  // global_addr [120:96]
                | (2u << 30)};                    // type=2 ("image")
  v4i gz4 = {0, 0, 0, 0};
  v8i gz8 = {0, 0, 0, 0, 0, 0, 0, 0};
  __builtin_amdgcn_tensor_load_to_lds(g0, g1, gz4, gz4, gz8, 0);
}
#endif

// ---------------------------------------------------------------------------
// f32 -> bf16 conversion (elementwise)
// ---------------------------------------------------------------------------
__global__ void cvt_f32_bf16(const float* __restrict__ in,
                             __bf16* __restrict__ out, int n) {
  int i = blockIdx.x * blockDim.x + threadIdx.x;
  if (i < n) out[i] = (__bf16)in[i];
}

// ---------------------------------------------------------------------------
// f32 [K][N] -> bf16 [N][K] transposed conversion (tiled through LDS).
// ---------------------------------------------------------------------------
__global__ void cvt_transpose_bf16(const float* __restrict__ in,
                                   __bf16* __restrict__ out) {
  __shared__ float tile[32][33];
  const int n0 = blockIdx.x * 32;
  const int k0 = blockIdx.y * 32;
  const int tx = threadIdx.x & 31;
  const int ty = threadIdx.x >> 5;
#pragma unroll
  for (int i = 0; i < 32; i += 8)
    tile[ty + i][tx] = in[(size_t)(k0 + ty + i) * D_DIM + n0 + tx];
  __syncthreads();
#pragma unroll
  for (int i = 0; i < 32; i += 8)
    out[(size_t)(n0 + ty + i) * D_DIM + k0 + tx] = (__bf16)tile[tx][ty + i];
}

// ---------------------------------------------------------------------------
// GEMM mainloop: C[128x64] tile of A[M x K] @ Bt[N x K]^T, bf16 in, f32 acc.
// 128 threads = 4 waves; wave w computes rows [w*32, w*32+32) x 64 cols.
// Both tiles row-contiguous -> staged by TDM, double-buffered.
// acc = 8 x v8f = 64 VGPRs per thread: no spills.
// ---------------------------------------------------------------------------
#define LDS_A 56  // padded stride (elems): 112B rows, conflict-free banks

__device__ __forceinline__ void gemm_mainloop(const __bf16* __restrict__ A,
                                              const __bf16* __restrict__ Bt,
                                              int K, int lda, int ldb,
                                              int m0, int n0,
                                              __bf16 (*sA)[128 * LDS_A],
                                              __bf16 (*sBt)[64 * LDS_A],
                                              v8f acc[2][4]) {
  const int t = threadIdx.x;
  const int w = t >> 5;
  const v8f vzero = {0.f, 0.f, 0.f, 0.f, 0.f, 0.f, 0.f, 0.f};
#pragma unroll
  for (int ai = 0; ai < 2; ++ai)
#pragma unroll
    for (int j = 0; j < 4; ++j) acc[ai][j] = vzero;

#if HAVE_TDM
  const unsigned sA_lds = (unsigned)(unsigned long long)&sA[0][0];
  const unsigned sB_lds = (unsigned)(unsigned long long)&sBt[0][0];
  const unsigned ABUF   = 128u * LDS_A * 2u;
  const unsigned BBUF   = 64u * LDS_A * 2u;
  // 64B rows (interval code 3), pad 12 dwords (code 11) -> LDS stride 56.
  const v8i g1a = tdm_make_g1(32, 128, (unsigned long long)lda, 3, 11);
  const v8i g1b = tdm_make_g1(32, 64,  (unsigned long long)ldb, 3, 11);
  if (t < 32) {
    tdm_issue(sA_lds, A  + (size_t)m0 * lda, g1a);
    tdm_issue(sB_lds, Bt + (size_t)n0 * ldb, g1b);
    __builtin_amdgcn_s_wait_tensorcnt(0);
  }
  __syncthreads();
#endif

  int cur = 0;
  for (int k0 = 0; k0 < K; k0 += 32) {
#if HAVE_TDM
    if (k0 + 32 < K && t < 32) {
      tdm_issue(sA_lds + (unsigned)(cur ^ 1) * ABUF,
                A + (size_t)m0 * lda + k0 + 32, g1a);
      tdm_issue(sB_lds + (unsigned)(cur ^ 1) * BBUF,
                Bt + (size_t)n0 * ldb + k0 + 32, g1b);
    }
#else
    __syncthreads();
#pragma unroll
    for (int i = 0; i < 4; ++i) {  // A tile: 4096 elems, 8 per thread per iter
      int e = i * 1024 + t * 8;
      int r = e >> 5, c = e & 31;
      uint4 da = *(const uint4*)(A + (size_t)(m0 + r) * lda + k0 + c);
      unsigned int* pa = (unsigned int*)(&sA[0][0] + r * LDS_A + c);
      pa[0] = da.x; pa[1] = da.y; pa[2] = da.z; pa[3] = da.w;
    }
#pragma unroll
    for (int i = 0; i < 2; ++i) {  // B tile: 2048 elems
      int e = i * 1024 + t * 8;
      int r = e >> 5, c = e & 31;
      uint4 db = *(const uint4*)(Bt + (size_t)(n0 + r) * ldb + k0 + c);
      unsigned int* pb = (unsigned int*)(&sBt[0][0] + r * LDS_A + c);
      pb[0] = db.x; pb[1] = db.y; pb[2] = db.z; pb[3] = db.w;
    }
    __syncthreads();
#endif

    // ---- compute on current buffer: 2 A-frags x 4 B-frags = 8 WMMA ----
    v16bf af0 = load_frag(&sA[cur][0], w * 32,      LDS_A, 0);
    v16bf af1 = load_frag(&sA[cur][0], w * 32 + 16, LDS_A, 0);
#pragma unroll
    for (int j = 0; j < 4; ++j) {
      v16bf bf = load_frag(&sBt[cur][0], j * 16, LDS_A, 0);
      acc[0][j] = wmma_bf16(af0, bf, acc[0][j]);
      acc[1][j] = wmma_bf16(af1, bf, acc[1][j]);
    }

#if HAVE_TDM
    if (t < 32 && k0 + 32 < K) __builtin_amdgcn_s_wait_tensorcnt(0);
    __syncthreads();
    cur ^= 1;
#endif
  }
}

// ---------------------------------------------------------------------------
// Fused Q/K/V projection: z = blockIdx.z selects weight + destination.
// Q scattered to [B][H][S][HD] pre-scaled by 1/sqrt(HD)=0.125; K likewise;
// V written TRANSPOSED to [B][H][HD][S] so attention can TDM-stage it.
// ---------------------------------------------------------------------------
__global__ void __launch_bounds__(128)
qkv_gemm_kernel(const __bf16* __restrict__ xb,
                const __bf16* __restrict__ Wtb,  // 3x[N][K]
                __bf16* __restrict__ qkv) {
  __shared__ __bf16 sA[2][128 * LDS_A];
  __shared__ __bf16 sBt[2][64 * LDS_A];
  const int m0 = blockIdx.x * 128;
  const int n0 = blockIdx.y * 64;
  const int z  = blockIdx.z;
  const __bf16* Wt  = Wtb + (size_t)z * D_DIM * D_DIM;
  __bf16*       out = qkv + (size_t)z * (NB * NH * S_LEN * HD);
  const float scale = (z == 0) ? 0.125f : 1.0f;

  v8f acc[2][4];
  gemm_mainloop(xb, Wt, D_DIM, D_DIM, D_DIM, m0, n0, sA, sBt, acc);

  const int lane = threadIdx.x & 31;
  const int w    = threadIdx.x >> 5;
  const int hi   = lane >> 4;
  const int nl   = lane & 15;
#pragma unroll
  for (int ai = 0; ai < 2; ++ai)
#pragma unroll
    for (int j = 0; j < 4; ++j) {
      int n = n0 + j * 16 + nl;
      int h = n >> 6, hd = n & (HD - 1);
#pragma unroll
      for (int v = 0; v < 8; ++v) {
        int m = m0 + w * 32 + ai * 16 + hi * 8 + v;  // C row = v + 8*(lane>=16)
        int b = m >> 11, s = m & (S_LEN - 1);
        __bf16 val = (__bf16)(acc[ai][j][v] * scale);
        if (z == 2)  // V transposed: [B][H][HD][S]
          out[((size_t)(b * NH + h) * HD + hd) * S_LEN + s] = val;
        else         // Q/K: [B][H][S][HD]
          out[(((size_t)(b * NH + h) * S_LEN + s) << 6) + hd] = val;
      }
    }
}

// ---------------------------------------------------------------------------
// Output projection: out[f32] = ctx_bf16 @ Wo^T(pre-transposed) + bo
// ---------------------------------------------------------------------------
__global__ void __launch_bounds__(128)
out_gemm_kernel(const __bf16* __restrict__ ctxb,
                const __bf16* __restrict__ Wot,
                const float* __restrict__ bo,
                float* __restrict__ out) {
  __shared__ __bf16 sA[2][128 * LDS_A];
  __shared__ __bf16 sBt[2][64 * LDS_A];
  const int m0 = blockIdx.x * 128;
  const int n0 = blockIdx.y * 64;

  v8f acc[2][4];
  gemm_mainloop(ctxb, Wot, D_DIM, D_DIM, D_DIM, m0, n0, sA, sBt, acc);

  const int lane = threadIdx.x & 31;
  const int w    = threadIdx.x >> 5;
  const int hi   = lane >> 4;
  const int nl   = lane & 15;
#pragma unroll
  for (int ai = 0; ai < 2; ++ai)
#pragma unroll
    for (int j = 0; j < 4; ++j) {
      int n = n0 + j * 16 + nl;
      float bias = bo[n];
#pragma unroll
      for (int v = 0; v < 8; ++v) {
        int m = m0 + w * 32 + ai * 16 + hi * 8 + v;
        out[(size_t)m * D_DIM + n] = acc[ai][j][v] + bias;
      }
    }
}

// ---------------------------------------------------------------------------
// Flash attention: one WG per (b, h, 64-row q block); 4 waves x 16 q rows.
// K tile [key][hd] and V tile [hd][key] both row-contiguous in memory
// (V pre-transposed by projection) -> both staged by TDM, double-buffered.
// ---------------------------------------------------------------------------
#define LDS_K 72  // padded stride: 128B row + 16B pad (4 dwords), bank-clean
#define LDS_P 72

__global__ void __launch_bounds__(128)
attn_kernel(const __bf16* __restrict__ Qm,
            const __bf16* __restrict__ Km,   // [B][H][S][HD]
            const __bf16* __restrict__ Vt,   // [B][H][HD][S]
            __bf16* __restrict__ ctx) {
  __shared__ __bf16 sK[2][64 * LDS_K];    // [key][hd]
  __shared__ __bf16 sVt[2][64 * LDS_K];   // [hd][key]
  __shared__ __bf16 sP[4][16 * LDS_P];    // per-wave P tile [qrow][key]

  const int qblk = blockIdx.x;
  const int h    = blockIdx.y;
  const int b    = blockIdx.z;
  const int t    = threadIdx.x;
  const int w    = t >> 5;
  const int lane = t & 31;
  const int hi   = lane >> 4;
  const int nl   = lane & 15;

  const size_t bh = ((size_t)b * NH + h) * S_LEN * HD;  // same for K and Vt
  const int q0 = qblk * 64 + w * 16;      // first q row of this wave (in S)

  // Q fragments held in registers for the whole kernel (pre-scaled by 0.125).
  v16bf qf0 = load_frag(Qm + bh, q0, HD, 0);   // hd 0..31
  v16bf qf1 = load_frag(Qm + bh, q0, HD, 32);  // hd 32..63

  const v8f vzero = {0.f, 0.f, 0.f, 0.f, 0.f, 0.f, 0.f, 0.f};
  v8f cacc[4];
#pragma unroll
  for (int j = 0; j < 4; ++j) cacc[j] = vzero;
  float m_run[8], l_run[8];
#pragma unroll
  for (int v = 0; v < 8; ++v) { m_run[v] = -3.0e38f; l_run[v] = 0.0f; }

#if HAVE_TDM
  const unsigned sK_lds = (unsigned)(unsigned long long)&sK[0][0];
  const unsigned sV_lds = (unsigned)(unsigned long long)&sVt[0][0];
  const unsigned KBUF   = 64u * LDS_K * 2u;
  // 128B rows (interval code 4), pad 4 dwords (code 3) -> LDS stride 72.
  const v8i g1k = tdm_make_g1(64, 64, (unsigned long long)HD,    4, 3);
  const v8i g1v = tdm_make_g1(64, 64, (unsigned long long)S_LEN, 4, 3);
  if (t < 32) {
    tdm_issue(sK_lds, Km + bh, g1k);
    tdm_issue(sV_lds, Vt + bh, g1v);
    __builtin_amdgcn_s_wait_tensorcnt(0);
  }
  __syncthreads();
#endif

  int cur = 0;
  for (int kb = 0; kb <= qblk; ++kb) {
#if HAVE_TDM
    if (kb < qblk && t < 32) {
      tdm_issue(sK_lds + (unsigned)(cur ^ 1) * KBUF,
                Km + bh + (size_t)(kb + 1) * 64 * HD, g1k);
      tdm_issue(sV_lds + (unsigned)(cur ^ 1) * KBUF,
                Vt + bh + (size_t)(kb + 1) * 64, g1v);
    }
    // Warm L2 for the block after the one in flight.
    if (kb + 2 <= qblk) {
      __builtin_prefetch(Km + bh + (size_t)(kb + 2) * 64 * HD + t * 32, 0, 1);
      __builtin_prefetch(Vt + bh + (size_t)(t & 63) * S_LEN + (kb + 2) * 64,
                         0, 1);
    }
#else
    __syncthreads();
#pragma unroll
    for (int i = 0; i < 4; ++i) {
      int e = i * 1024 + t * 8;
      int r = e >> 6, c = e & 63;
      uint4 dk = *(const uint4*)(Km + bh + (size_t)kb * 64 * HD + r * HD + c);
      unsigned int* pk = (unsigned int*)(&sK[0][0] + r * LDS_K + c);
      pk[0] = dk.x; pk[1] = dk.y; pk[2] = dk.z; pk[3] = dk.w;
      uint4 dv = *(const uint4*)(Vt + bh + (size_t)r * S_LEN + kb * 64 + c);
      unsigned int* pv = (unsigned int*)(&sVt[0][0] + r * LDS_K + c);
      pv[0] = dv.x; pv[1] = dv.y; pv[2] = dv.z; pv[3] = dv.w;
    }
    __syncthreads();
#endif

    // scores S = Q @ K^T : 4 key subtiles x (2 WMMA over hd).
    v8f sc[4];
#pragma unroll
    for (int j = 0; j < 4; ++j) {
      v16bf kfa = load_frag(&sK[cur][0], j * 16, LDS_K, 0);
      v8f z = wmma_bf16(qf0, kfa, vzero);
      v16bf kfb = load_frag(&sK[cur][0], j * 16, LDS_K, 32);
      sc[j] = wmma_bf16(qf1, kfb, z);
    }

    // Causal mask needed only on the diagonal key block.
    if (kb == qblk) {
#pragma unroll
      for (int j = 0; j < 4; ++j) {
        int key = kb * 64 + j * 16 + nl;
#pragma unroll
        for (int v = 0; v < 8; ++v) {
          int q = q0 + hi * 8 + v;
          if (key > q) sc[j][v] = -3.0e38f;
        }
      }
    }

    // Online softmax per C-layout row slot (row = v + 8*hi, 16-lane groups).
    float pl[4][8];
#pragma unroll
    for (int v = 0; v < 8; ++v) {
      float mx = fmaxf(fmaxf(sc[0][v], sc[1][v]), fmaxf(sc[2][v], sc[3][v]));
#pragma unroll
      for (int off = 1; off < 16; off <<= 1)
        mx = fmaxf(mx, __shfl_xor(mx, off, 32));
      float mnew  = fmaxf(m_run[v], mx);
      float alpha = __expf(m_run[v] - mnew);
      float rs = 0.0f;
#pragma unroll
      for (int j = 0; j < 4; ++j) {
        float p = __expf(sc[j][v] - mnew);
        pl[j][v] = p;
        rs += p;
      }
#pragma unroll
      for (int off = 1; off < 16; off <<= 1) rs += __shfl_xor(rs, off, 32);
      l_run[v] = l_run[v] * alpha + rs;
      m_run[v] = mnew;
#pragma unroll
      for (int j = 0; j < 4; ++j) cacc[j][v] *= alpha;
    }

    // Re-layout P: C layout -> A layout via per-wave LDS tile.
    __bf16* Pw = sP[w];
#pragma unroll
    for (int j = 0; j < 4; ++j)
#pragma unroll
      for (int v = 0; v < 8; ++v)
        Pw[(hi * 8 + v) * LDS_P + j * 16 + nl] = (__bf16)pl[j][v];
    asm volatile("s_wait_dscnt 0" ::: "memory");  // wave-local LDS RAW

    v16bf pf0 = load_frag(Pw, 0, LDS_P, 0);   // keys 0..31
    v16bf pf1 = load_frag(Pw, 0, LDS_P, 32);  // keys 32..63
#pragma unroll
    for (int j = 0; j < 4; ++j) {             // hd subtiles
      v16bf vf0 = load_frag(&sVt[cur][0], j * 16, LDS_K, 0);
      cacc[j] = wmma_bf16(pf0, vf0, cacc[j]);
      v16bf vf1 = load_frag(&sVt[cur][0], j * 16, LDS_K, 32);
      cacc[j] = wmma_bf16(pf1, vf1, cacc[j]);
    }

#if HAVE_TDM
    if (t < 32 && kb < qblk) __builtin_amdgcn_s_wait_tensorcnt(0);
    __syncthreads();
    cur ^= 1;
#endif
  }

  // Normalize and scatter ctx back to [B*S][D] bf16 for the output GEMM.
#pragma unroll
  for (int j = 0; j < 4; ++j) {
#pragma unroll
    for (int v = 0; v < 8; ++v) {
      float val = cacc[j][v] / l_run[v];
      int row = b * S_LEN + q0 + hi * 8 + v;
      int col = h * HD + j * 16 + nl;
      ctx[(size_t)row * D_DIM + col] = (__bf16)val;
    }
  }
}

// ---------------------------------------------------------------------------
// Host-side launch
// ---------------------------------------------------------------------------
extern "C" void kernel_launch(void* const* d_in, const int* in_sizes, int n_in,
                              void* d_out, int out_size, void* d_ws, size_t ws_size,
                              hipStream_t stream) {
  const float* x  = (const float*)d_in[0];
  const float* Wq = (const float*)d_in[1];
  const float* Wk = (const float*)d_in[2];
  const float* Wv = (const float*)d_in[3];
  const float* Wo = (const float*)d_in[4];
  const float* bo = (const float*)d_in[5];

  const size_t MB = 1024ull * 1024ull;
  char* ws = (char*)d_ws;
  __bf16* xb   = (__bf16*)(ws);             // [4096][1024]               8 MB
  __bf16* Wtb  = (__bf16*)(ws + 8  * MB);   // Wq,Wk,Wv,Wo transposed     8 MB
  __bf16* qkv  = (__bf16*)(ws + 16 * MB);   // Q,K [B][H][S][HD]; Vt     24 MB
  __bf16* ctxb = (__bf16*)(ws + 40 * MB);   // [4096][1024]               8 MB

  const int NX = NB * S_LEN * D_DIM;        // 4194304
  const int NW = D_DIM * D_DIM;             // 1048576

  cvt_f32_bf16<<<NX / 256, 256, 0, stream>>>(x, xb, NX);
  dim3 tg(D_DIM / 32, D_DIM / 32);
  cvt_transpose_bf16<<<tg, 256, 0, stream>>>(Wq, Wtb + 0ull * NW);
  cvt_transpose_bf16<<<tg, 256, 0, stream>>>(Wk, Wtb + 1ull * NW);
  cvt_transpose_bf16<<<tg, 256, 0, stream>>>(Wv, Wtb + 2ull * NW);
  cvt_transpose_bf16<<<tg, 256, 0, stream>>>(Wo, Wtb + 3ull * NW);

  // Q/K/V projections: M=4096, N=1024, K=1024; z selects Q/K/V.
  qkv_gemm_kernel<<<dim3(NX / D_DIM / 128, D_DIM / 64, 3), 128, 0, stream>>>(
      xb, Wtb, qkv);

  const size_t qkvN = (size_t)NB * NH * S_LEN * HD;  // elements per tensor
  attn_kernel<<<dim3(S_LEN / 64, NH, NB), 128, 0, stream>>>(
      qkv, qkv + qkvN, qkv + 2 * qkvN, ctxb);

  out_gemm_kernel<<<dim3(NX / D_DIM / 128, D_DIM / 64), 128, 0, stream>>>(
      ctxb, Wtb + 3ull * NW, bo, (float*)d_out);
}
